// BCellIRTActor_51702816309840
// MI455X (gfx1250) — matrix operand order repaired and checked by reading, not compiled, not run on hardware
//
#include <hip/hip_runtime.h>
#include <hip/hip_bf16.h>
#include <math.h>

typedef __attribute__((ext_vector_type(16))) _Float16 v16h;
typedef __attribute__((ext_vector_type(8)))  float    v8f;

#define WMMA_F16(a,b,c) __builtin_amdgcn_wmma_f32_16x16x32_f16(false,(a),false,(b),(short)0,(c),false,false)

// ---------------------------------------------------------------------------
// Fast branch-free math: native v_exp_f32 / v_rcp_f32 / v_rsq_f32 / v_sqrt_f32
// (no libm EXEC divergence, no IEEE div_scale/div_fixup sequences)
// ---------------------------------------------------------------------------
__device__ __forceinline__ float fast_rcp(float x)  { return __builtin_amdgcn_rcpf(x); }
__device__ __forceinline__ float fast_rsq(float x)  { return __builtin_amdgcn_rsqf(x); }
__device__ __forceinline__ float fast_sqrt(float x) { return __builtin_amdgcn_sqrtf(x); }

__device__ __forceinline__ float fast_tanh(float x) {
    float ax = fabsf(x);
    float t  = __expf(-2.f * ax);                    // (0,1], ->0 for large |x|
    float r  = (1.f - t) * fast_rcp(1.f + t);        // tanh(|x|), saturates to 1
    return copysignf(r, x);
}

__device__ __forceinline__ float fast_sigmoid(float x) {
    return fast_rcp(1.f + __expf(-x));
}

// ---------------------------------------------------------------------------
// WMMA fragment helpers
// ---------------------------------------------------------------------------

// A-fragment (16x32 f16, M x K) from LDS row-major f16 [row*strideH + k].
// Layout: lanes 0-15 -> M=lane, halves 0..15 = K kbase..kbase+15
//         lanes 16-31 -> M=lane-16, halves 0..15 = K kbase+16..kbase+31 (matches B prep).
__device__ __forceinline__ v16h lds_a_frag(const _Float16* base, int strideH, int kbase, int lane) {
    int m  = lane & 15;
    int hi = lane >> 4;           // 0 or 1
    const _Float16* p = base + m * strideH + kbase + hi * 16;
    union { v16h v; uint32_t u[8]; } f;
    const uint32_t* p0 = (const uint32_t*)p;
    const uint32_t* p1 = (const uint32_t*)(p + 8);
    f.u[0] = p0[0]; f.u[1] = p0[1]; f.u[2] = p0[2]; f.u[3] = p0[3];
    f.u[4] = p1[0]; f.u[5] = p1[1]; f.u[6] = p1[2]; f.u[7] = p1[3];
    return f.v;
}

// Butterfly reductions within each 16-lane group (lanes 0-15 and 16-31 reduce independently).
__device__ __forceinline__ float bf_sum16(float x) {
    x += __shfl_xor(x, 1, 32); x += __shfl_xor(x, 2, 32);
    x += __shfl_xor(x, 4, 32); x += __shfl_xor(x, 8, 32);
    return x;
}
__device__ __forceinline__ float bf_max16(float x) {
    x = fmaxf(x, __shfl_xor(x, 1, 32)); x = fmaxf(x, __shfl_xor(x, 2, 32));
    x = fmaxf(x, __shfl_xor(x, 4, 32)); x = fmaxf(x, __shfl_xor(x, 8, 32));
    return x;
}

// ---------------------------------------------------------------------------
// Setup kernels: convert weights to WMMA B-fragment-ready f16 layouts.
// Layout: dst[((tile*nk + ks)*32 + lane)*16 + i]
//   n = tile*16 + (lane&15); k = ks*32 + ((lane&16)?16:0) + i  (i = 0..15)
// so each lane's 16 halves are one contiguous 32-byte chunk at runtime.
// ---------------------------------------------------------------------------
__global__ void k_prep_b(const float* __restrict__ src, _Float16* __restrict__ dst,
                         int K, int N, int nk, int nt, int transpose, int srcLd) {
    int idx = blockIdx.x * blockDim.x + threadIdx.x;
    int total = nt * nk * 512;
    if (idx >= total) return;
    int i    = idx & 15;
    int lane = (idx >> 4) & 31;
    int tk   = idx >> 9;
    int ks   = tk % nk;
    int tile = tk / nk;
    int n = tile * 16 + (lane & 15);
    int k = ks * 32 + ((lane & 16) ? 16 : 0) + i;
    float v = 0.f;
    if (k < K && n < N)
        v = transpose ? src[(size_t)n * srcLd + k] : src[(size_t)k * srcLd + n];
    dst[idx] = (_Float16)v;
}

// W2pk[k, tok*16+n] = sum_e enc_W2[k, tok*256+e] * pk[n, e]   (256 x 128), B-frag layout
__global__ void k_prep_w2pk(const float* __restrict__ W2, const float* __restrict__ pk,
                            _Float16* __restrict__ dst) {
    int idx = blockIdx.x * blockDim.x + threadIdx.x;   // 8*8*512 = 32768
    if (idx >= 8 * 8 * 512) return;
    int i    = idx & 15;
    int lane = (idx >> 4) & 31;
    int tk   = idx >> 9;
    int ks   = tk % 8;
    int tok  = tk / 8;             // tile == token
    int n    = lane & 15;
    int k    = ks * 32 + ((lane & 16) ? 16 : 0) + i;   // 0..255
    const float* w2row = W2 + (size_t)k * 2048 + (size_t)tok * 256;
    const float* pkrow = pk + (size_t)n * 256;
    float s = 0.f;
    for (int e = 0; e < 256; e++) s += w2row[e] * pkrow[e];
    dst[idx] = (_Float16)s;
}

// b2pk[tok*16+n] = sum_e enc_b2[tok*256+e] * pk[n,e]
__global__ void k_b2pk(const float* __restrict__ b2, const float* __restrict__ pk,
                       float* __restrict__ dst) {
    int idx = threadIdx.x;          // 128
    if (idx >= 128) return;
    int tok = idx >> 4, n = idx & 15;
    float s = 0.f;
    for (int e = 0; e < 256; e++) s += b2[tok * 256 + e] * pk[n * 256 + e];
    dst[idx] = s;
}

// Decoder (B-independent): hd = relu(pk@dec_W1 + b1); conc = softplus(hd@dec_W2 + b2)
__global__ void k_dec(const float* __restrict__ pk, const float* __restrict__ dW1,
                      const float* __restrict__ db1, const float* __restrict__ dW2,
                      const float* __restrict__ db2, float* __restrict__ conc) {
    __shared__ float hd[16 * 128];
    int tid = threadIdx.x;          // 256
    for (int o = tid; o < 2048; o += 256) {
        int m = o >> 7, hh = o & 127;
        float s = db1[m * 128 + hh];
        for (int e = 0; e < 256; e++)
            s += pk[m * 256 + e] * dW1[((size_t)m * 256 + e) * 128 + hh];
        hd[o] = fmaxf(s, 0.f);
    }
    __syncthreads();
    for (int o = tid; o < 480; o += 256) {
        int m = o / 30, a = o % 30;
        float s = db2[m * 30 + a];
        for (int h = 0; h < 128; h++)
            s += hd[m * 128 + h] * dW2[((size_t)m * 128 + h) * 30 + a];
        // softplus: log1p(exp(s)), stable for large s
        conc[o] = (s > 20.f) ? s : log1pf(__expf(s));
    }
}

// ---------------------------------------------------------------------------
// Main fused kernel: 64 threads = 2 waves, each wave owns 16 rows of B.
// ---------------------------------------------------------------------------
#define ST_STRIDE 328   // 320 + 8 pad (halves)
#define MF_STRIDE 36
#define DH_STRIDE 264   // 256 + 8 pad

__global__ __launch_bounds__(64) void k_main(
    const float* __restrict__ state, const float* __restrict__ fitness,
    const float* __restrict__ enc_b1, const float* __restrict__ ln_g, const float* __restrict__ ln_b,
    const float* __restrict__ tc_b, const float* __restrict__ tc_cW, const float* __restrict__ tc_cb,
    const float* __restrict__ w_prev,
    const _Float16* __restrict__ w1p, const _Float16* __restrict__ w2pkp,
    const _Float16* __restrict__ tcwp, const _Float16* __restrict__ pktp,
    const float* __restrict__ b2pk, const float* __restrict__ conc,
    float* __restrict__ out) {
    __shared__ _Float16 sState[2][16 * ST_STRIDE];
    __shared__ _Float16 sMf[2][16 * MF_STRIDE];
    __shared__ _Float16 sDan[2][16 * DH_STRIDE];
    __shared__ _Float16 sH[2][16 * DH_STRIDE];
    __shared__ float    sW[2][256];

    const int tid = threadIdx.x;
    const int wid = tid >> 5;
    const int lane = tid & 31;
    const int nl = lane & 15;
    const int mrow = (lane >> 4) * 8;    // row offset for this half-wave in C/D layout
    const int rowBase = blockIdx.x * 32 + wid * 16;

    // --- 1. state tile -> LDS f16 (coalesced float4 loads; 300 = 75 float4) ---
    for (int r = 0; r < 16; r++) {
        const float4* src = (const float4*)(state + (size_t)(rowBase + r) * 300);
        for (int c4 = lane; c4 < 75; c4 += 32) {
            float4 v = src[c4];
            union { _Float16 h[4]; uint2 u; } p4;
            p4.h[0] = (_Float16)v.x; p4.h[1] = (_Float16)v.y;
            p4.h[2] = (_Float16)v.z; p4.h[3] = (_Float16)v.w;
            *(uint2*)&sState[wid][r * ST_STRIDE + c4 * 4] = p4.u;
        }
        if (lane < 28) sState[wid][r * ST_STRIDE + 300 + lane] = (_Float16)0.f;
    }

    // --- 2. features (f32 from global), mf -> LDS f16 padded K 12->32 ---
    if (lane < 16) {
        const float* s = state + (size_t)(rowBase + lane) * 300;
        float bal = s[0];
        float ps = 0.f, pq = 0.f, tv = 0.f;
        for (int i = 0; i < 30; i++) {
            float p = s[1 + i];
            ps += p; pq += p * p; tv += p * s[31 + i];
        }
        float pm  = ps * (1.f / 30.f);
        float var = (pq - 30.f * pm * pm) * (1.f / 29.f);   // ddof=1
        float sd  = fast_sqrt(fmaxf(var, 0.f)) + 1e-8f;
        tv += bal;
        float cr = bal * fast_rcp(tv + 1e-8f);
        float mf[12];
        mf[0] = bal; mf[1] = pm; mf[2] = sd; mf[3] = cr;
        for (int i = 0; i < 8; i++) mf[4 + i] = s[61 + 30 * i];
        for (int i = 0; i < MF_STRIDE; i++)
            sMf[wid][lane * MF_STRIDE + i] = (_Float16)(i < 12 ? mf[i] : 0.f);
    }
    __syncthreads();

    // --- 3. danger = tanh(mf@tcW + tc_b); crisis -> alpha ---
    float alpha[8];
    {
        v16h a = lds_a_frag(&sMf[wid][0], MF_STRIDE, 0, lane);
        float cdot[8];
        #pragma unroll
        for (int j = 0; j < 8; j++) cdot[j] = 0.f;
        for (int t = 0; t < 16; t++) {
            v16h b = *(const v16h*)(tcwp + ((size_t)t * 32 + lane) * 16);
            v8f c = {};
            c = WMMA_F16(a, b, c);
            float tb = tc_b[t * 16 + nl];
            float cw = tc_cW[t * 16 + nl];
            #pragma unroll
            for (int j = 0; j < 8; j++) {
                float x = fast_tanh(c[j] + tb);
                cdot[j] += x * cw;
                sDan[wid][(j + mrow) * DH_STRIDE + t * 16 + nl] = (_Float16)x;
            }
        }
        float cb = tc_cb[0];
        #pragma unroll
        for (int j = 0; j < 8; j++) {
            float d = bf_sum16(cdot[j]);
            float crisis = fast_sigmoid(d + cb);
            alpha[j] = 0.06f + 0.24f * (1.f - crisis);    // ALPHA_MIN + (MAX-MIN)*(1-crisis)
        }
    }
    __syncthreads();

    // --- 4. dlog = danger @ pk^T  (16x16 tile, kept in C layout) ---
    v8f dlog = {};
    for (int ks = 0; ks < 8; ks++) {
        v16h a = lds_a_frag(&sDan[wid][0], DH_STRIDE, ks * 32, lane);
        v16h b = *(const v16h*)(pktp + ((size_t)ks * 32 + lane) * 16);
        dlog = WMMA_F16(a, b, dlog);
    }

    // --- 5. GEMM1: h_pre = state @ enc_W1  (16x256, 10 ksteps x 16 ntiles) ---
    v8f hacc[16];
    #pragma unroll
    for (int t = 0; t < 16; t++) { v8f z = {}; hacc[t] = z; }
    for (int ks = 0; ks < 10; ks++) {
        v16h a = lds_a_frag(&sState[wid][0], ST_STRIDE, ks * 32, lane);
        #pragma unroll
        for (int t = 0; t < 16; t++) {
            v16h b = *(const v16h*)(w1p + (((size_t)t * 10 + ks) * 32 + lane) * 16);
            hacc[t] = WMMA_F16(a, b, hacc[t]);
        }
    }

    // --- 6. bias + LayerNorm + ReLU -> sH (f16) ---
    {
        float s1[8], s2[8];
        #pragma unroll
        for (int j = 0; j < 8; j++) { s1[j] = 0.f; s2[j] = 0.f; }
        #pragma unroll
        for (int t = 0; t < 16; t++) {
            float bv = enc_b1[t * 16 + nl];
            #pragma unroll
            for (int j = 0; j < 8; j++) {
                float x = hacc[t][j] + bv;
                hacc[t][j] = x;
                s1[j] += x; s2[j] += x * x;
            }
        }
        float mu[8], ri[8];
        #pragma unroll
        for (int j = 0; j < 8; j++) {
            float a1 = bf_sum16(s1[j]);
            float a2 = bf_sum16(s2[j]);
            float m_ = a1 * (1.f / 256.f);
            float v_ = a2 * (1.f / 256.f) - m_ * m_;
            mu[j] = m_;
            ri[j] = fast_rsq(v_ + 1e-5f);
        }
        #pragma unroll
        for (int t = 0; t < 16; t++) {
            float g = ln_g[t * 16 + nl], bb = ln_b[t * 16 + nl];
            #pragma unroll
            for (int j = 0; j < 8; j++) {
                float y = (hacc[t][j] - mu[j]) * ri[j] * g + bb;
                y = fmaxf(y, 0.f);
                sH[wid][(j + mrow) * DH_STRIDE + t * 16 + nl] = (_Float16)y;
            }
        }
    }
    __syncthreads();

    // --- 7. GEMM2: L = h @ W2pk  (16x128, 8 ksteps x 8 ntiles; tile t == token) ---
    v8f lacc[8];
    #pragma unroll
    for (int t = 0; t < 8; t++) { v8f z = {}; lacc[t] = z; }
    for (int ks = 0; ks < 8; ks++) {
        v16h a = lds_a_frag(&sH[wid][0], DH_STRIDE, ks * 32, lane);
        #pragma unroll
        for (int t = 0; t < 8; t++) {
            v16h b = *(const v16h*)(w2pkp + (((size_t)t * 8 + ks) * 32 + lane) * 16);
            lacc[t] = WMMA_F16(a, b, lacc[t]);
        }
    }

    // --- 8. logits -> per-token softmax over 16 protos -> w_ot ---
    const float scale = 0.0625f;   // EMB^-0.5 = 1/16
    float wot[8];
    #pragma unroll
    for (int j = 0; j < 8; j++) wot[j] = 0.f;
    #pragma unroll
    for (int t = 0; t < 8; t++) {
        float bp = b2pk[t * 16 + nl];
        #pragma unroll
        for (int j = 0; j < 8; j++) {
            float x = (lacc[t][j] + bp) * scale + dlog[j] * scale;
            float mx = bf_max16(x);
            float e = __expf(x - mx);
            float ssum = bf_sum16(e);
            wot[j] += e * fast_rcp(ssum) * 0.125f;   // mean over 8 tokens
        }
    }

    // --- 9. w_rep, blend, normalize -> sW ---
    #pragma unroll
    for (int j = 0; j < 8; j++) {
        int m = j + mrow;
        int row = rowBase + m;
        float f = fitness[(size_t)row * 16 + nl];
        float wr = w_prev[nl] * __expf(0.1f * f);
        float s = bf_sum16(wr);
        wr *= fast_rcp(s + 1e-8f);
        float w = alpha[j] * wot[j] + (1.f - alpha[j]) * wr;
        float s2 = bf_sum16(w);
        w *= fast_rcp(s2 + 1e-8f);
        sW[wid][m * 16 + nl] = w;
    }
    __syncthreads();

    // --- 10. tail: mixed = w@conc + 1; softmax; clip; renorm ---
    if (lane < 16) {
        int m = lane;
        int row = rowBase + m;
        float wv[16];
        #pragma unroll
        for (int n = 0; n < 16; n++) wv[n] = sW[wid][m * 16 + n];
        float mixed[30];
        float mx = -1e30f;
        for (int a = 0; a < 30; a++) {
            float s = 1.f;
            #pragma unroll
            for (int n = 0; n < 16; n++) s += wv[n] * conc[n * 30 + a];
            mixed[a] = s;
            mx = fmaxf(mx, s);
        }
        float S = 0.f;
        for (int a = 0; a < 30; a++) { mixed[a] = __expf(mixed[a] - mx); S += mixed[a]; }
        float inv = fast_rcp(S);
        float S2 = 0.f;
        for (int a = 0; a < 30; a++) {
            float p = fminf(fmaxf(mixed[a] * inv, 0.f), 1.f);
            mixed[a] = p; S2 += p;
        }
        float inv2 = fast_rcp(S2 + 1e-8f);
        for (int a = 0; a < 30; a++) out[(size_t)row * 30 + a] = mixed[a] * inv2;
    }
}

// ---------------------------------------------------------------------------
// Host launcher
// ---------------------------------------------------------------------------
extern "C" void kernel_launch(void* const* d_in, const int* in_sizes, int n_in,
                              void* d_out, int out_size, void* d_ws, size_t ws_size,
                              hipStream_t stream) {
    const float* state   = (const float*)d_in[0];
    const float* fitness = (const float*)d_in[1];
    const float* pk      = (const float*)d_in[2];
    const float* W1      = (const float*)d_in[3];
    const float* b1      = (const float*)d_in[4];
    const float* lng     = (const float*)d_in[5];
    const float* lnb     = (const float*)d_in[6];
    const float* W2      = (const float*)d_in[7];
    const float* b2      = (const float*)d_in[8];
    const float* dW1     = (const float*)d_in[9];
    const float* db1     = (const float*)d_in[10];
    const float* dW2     = (const float*)d_in[11];
    const float* db2     = (const float*)d_in[12];
    const float* tcW     = (const float*)d_in[13];
    const float* tcb     = (const float*)d_in[14];
    const float* tccW    = (const float*)d_in[15];
    const float* tccb    = (const float*)d_in[16];
    const float* wprev   = (const float*)d_in[19];
    float* out = (float*)d_out;

    char* ws = (char*)d_ws;
    float*    b2pk  = (float*)ws;                                   // 128 f
    float*    conc  = (float*)(ws + 512);                           // 480 f
    _Float16* w1p   = (_Float16*)(ws + 4096);                       // 16nt*10ks*512 = 81920 h
    _Float16* w2pkp = (_Float16*)(ws + 4096 + 163840);              // 8*8*512 = 32768 h
    _Float16* tcwp  = (_Float16*)(ws + 4096 + 163840 + 65536);      // 16*1*512 = 8192 h
    _Float16* pktp  = (_Float16*)(ws + 4096 + 163840 + 65536 + 16384); // 1*8*512 = 4096 h

    // weight prep (tiny, stream-ordered before main)
    k_prep_b<<<320, 256, 0, stream>>>(W1, w1p, 300, 256, 10, 16, 0, 256);
    k_prep_b<<<32, 256, 0, stream>>>(tcW, tcwp, 12, 256, 1, 16, 0, 256);
    k_prep_b<<<16, 256, 0, stream>>>(pk, pktp, 256, 16, 8, 1, 1, 256);  // pk^T
    k_prep_w2pk<<<128, 256, 0, stream>>>(W2, pk, w2pkp);
    k_b2pk<<<1, 128, 0, stream>>>(b2, pk, b2pk);
    k_dec<<<1, 256, 0, stream>>>(pk, dW1, db1, dW2, db2, conc);

    // fused main pass: 16384 rows / 32 rows-per-block = 512 blocks of 64 threads
    k_main<<<512, 64, 0, stream>>>(state, fitness, b1, lng, lnb, tcb, tccW, tccb, wprev,
                                   w1p, w2pkp, tcwp, pktp, b2pk, conc, out);
}
